// CRTransNetBlock_87119116632159
// MI455X (gfx1250) — compile-verified, matrix operand order checked by reading, not run on hardware
//
#include <hip/hip_runtime.h>
#include <hip/hip_bf16.h>

// ---------------------------------------------------------------------------
// Problem constants (match reference)
// ---------------------------------------------------------------------------
#define BB 4
#define NT 1024
#define NS 1024
#define DD 512
#define EE 512
#define HH 8
#define HD 64
#define FF 2048
#define RH 16

typedef __attribute__((ext_vector_type(16))) _Float16 v16h;
typedef __attribute__((ext_vector_type(8)))  _Float16 v8h;
typedef __attribute__((ext_vector_type(8)))  float    v8f;
typedef __attribute__((ext_vector_type(4)))  int      v4i;

// Output tuple layout in d_out (floats): x, att, rel_emb, b_emb
#define OFF_X      0L
#define OFF_ATT    ((long)BB * NT * EE)                         // 2,097,152
#define OFF_RELEMB (OFF_ATT + (long)BB * HH * NT * NS)          // +33,554,432
#define OFF_BEMB   (OFF_RELEMB + (long)BB * NT * NS * RH)       // +67,108,864

// gfx1250 async global->LDS copy path (guarded; falls back to ds stores)
#if defined(__has_builtin)
#  if __has_builtin(__builtin_amdgcn_global_load_async_to_lds_b128) && \
      __has_builtin(__builtin_amdgcn_s_wait_asynccnt)
#    define ASYNC_OK 1
#  else
#    define ASYNC_OK 0
#  endif
#else
#  define ASYNC_OK 0
#endif

// Builtin signature (probe-confirmed by round-2 diagnostic):
//   void __builtin_amdgcn_global_load_async_to_lds_b128(
//       int4 addrspace(1)* gsrc, int4 addrspace(3)* ldst, imm i32 off, imm i32 cpol)
typedef __attribute__((address_space(1))) v4i gv4i;
typedef __attribute__((address_space(3))) v4i lv4i;

// Copy one 16-byte chunk global -> LDS (per lane).
__device__ inline void stage_b128(const _Float16* __restrict__ g,
                                  _Float16* __restrict__ l) {
#if ASYNC_OK
  __builtin_amdgcn_global_load_async_to_lds_b128(
      (gv4i*)(unsigned long long)(const void*)g,
      (lv4i*)(unsigned)(unsigned long long)(const void*)l, 0, 0);
#else
  *(v8h*)l = *(const v8h*)g;
#endif
}

__device__ inline void stage_wait() {
#if ASYNC_OK
  __builtin_amdgcn_s_wait_asynccnt(0);
#endif
}

// ---------------------------------------------------------------------------
// Conversion helpers
// ---------------------------------------------------------------------------
__global__ __launch_bounds__(256) void k_f2h(const float* __restrict__ in,
                                             _Float16* __restrict__ out, long n) {
  long i = (long)blockIdx.x * 256 + threadIdx.x;
  if (i < n) out[i] = (_Float16)in[i];
}

// in: f32 [Kd][Nd] row-major  ->  out: f16 [Nd][Kd] row-major ("Bt" layout)
__global__ __launch_bounds__(256) void k_transpose_f2h(const float* __restrict__ in,
                                                       _Float16* __restrict__ out,
                                                       int Kd, int Nd) {
  long i = (long)blockIdx.x * 256 + threadIdx.x;
  long total = (long)Kd * Nd;
  if (i >= total) return;
  int k = (int)(i % Kd);
  long n = i / Kd;
  out[i] = (_Float16)in[(long)k * Nd + n];
}

// ---------------------------------------------------------------------------
// WMMA fragment load: 16-bit A-matrix 16x32 layout (ISA 7.12.2).
// lanes 0-15: rows M=0..15, K-halves {0..7,16..23}; lanes 16-31: K {8..15,24..31}
// Works for global (global_load_b128) and LDS (ds_load_b128) pointers.
// ---------------------------------------------------------------------------
__device__ inline v16h load_frag(const _Float16* __restrict__ p) {
  v8h lo = *(const v8h*)p;
  v8h hi = *(const v8h*)(p + 16);
  v16h f;
#pragma unroll
  for (int j = 0; j < 8; ++j) { f[j] = lo[j]; f[8 + j] = hi[j]; }
  return f;
}

__device__ inline v8f wmma(v16h a, v16h b, v8f c) {
  return __builtin_amdgcn_wmma_f32_16x16x32_f16(false, a, false, b,
                                                (short)0, c, false, false);
}

// ---------------------------------------------------------------------------
// Generic batched GEMM:  D[z] = A[z] (MxK, f16) * Bt[z]^T (Bt is [N][K], f16)
// + bias, optional ReLU, optional f32 and/or f16 stores.
// Block = 128 threads (4 waves). Block tile: 128(M) x 32(N).
// Each wave: 32x32 tile = 4 accumulators, 4 WMMAs per K-step (fragment reuse).
// The 32x32 B tile is shared by all 4 waves -> staged in LDS (double-buffered)
// via gfx1250 async global->LDS copies when available.
// Requires: M % 128 == 0, N % 32 == 0, K % 32 == 0.
// ---------------------------------------------------------------------------
__global__ __launch_bounds__(128) void k_gemm_f16(
    const _Float16* __restrict__ A, long sA, int lda,
    const _Float16* __restrict__ Bt, long sB, int ldb,
    float* __restrict__ D, long sD, int ldd,
    _Float16* __restrict__ Dh, long sDh, int lddh,
    const float* __restrict__ bias, int M, int N, int K, int relu) {
  __shared__ alignas(16) _Float16 Bs[2][32 * 32];

  const int tid  = threadIdx.x;
  const int lane = tid & 31;
  const int wave = tid >> 5;
  const int z    = blockIdx.z;
  const int n0   = blockIdx.x * 32;
  const int m0   = blockIdx.y * 128 + wave * 32;
  const int r    = lane & 15;
  const int kg   = lane >> 4;

  const _Float16* Ab = A + (long)z * sA;
  const _Float16* Bb = Bt + (long)z * sB;

  // B staging: thread t handles 16B chunk t of the 32x32 tile (2 KB total).
  const int srow = tid >> 2;            // 0..31 (row of Bt tile)
  const int scol = (tid & 3) * 8;       // 0,8,16,24 (halfs)
  const _Float16* sg = Bb + (long)(n0 + srow) * ldb + scol;
  const int soff = srow * 32 + scol;

  const _Float16* ap0 = Ab + (long)(m0 + r) * lda + kg * 8;  // rows m0..m0+15
  const _Float16* ap1 = ap0 + 16 * lda;                      // rows m0+16..+31

  v8f c00 = {}, c01 = {}, c10 = {}, c11 = {};

  stage_b128(sg, &Bs[0][soff]);
  int cur = 0;
  for (int k0 = 0; k0 < K; k0 += 32) {
    stage_wait();
    __syncthreads();
    if (k0 + 32 < K) {
      stage_b128(sg + k0 + 32, &Bs[cur ^ 1][soff]);
      __builtin_prefetch(ap0 + k0 + 32, 0, 3);   // gfx1250 global_prefetch_b8
      __builtin_prefetch(ap1 + k0 + 32, 0, 3);
    }
    v16h a0 = load_frag(ap0 + k0);
    v16h a1 = load_frag(ap1 + k0);
    v16h b0 = load_frag(&Bs[cur][r * 32 + kg * 8]);          // Bt rows n0..+15
    v16h b1 = load_frag(&Bs[cur][(16 + r) * 32 + kg * 8]);   // Bt rows n0+16..+31
    c00 = wmma(a0, b0, c00);
    c01 = wmma(a0, b1, c01);
    c10 = wmma(a1, b0, c10);
    c11 = wmma(a1, b1, c11);
    cur ^= 1;
  }

  const int mA = m0 + kg * 8;
  const int mB = mA + 16;
  const int nA = n0 + r;
  const int nB = nA + 16;
  const float bvA = bias ? bias[nA] : 0.0f;
  const float bvB = bias ? bias[nB] : 0.0f;
#pragma unroll
  for (int i = 0; i < 8; ++i) {
    float x00 = c00[i] + bvA, x01 = c01[i] + bvB;
    float x10 = c10[i] + bvA, x11 = c11[i] + bvB;
    if (relu) {
      x00 = fmaxf(x00, 0.f); x01 = fmaxf(x01, 0.f);
      x10 = fmaxf(x10, 0.f); x11 = fmaxf(x11, 0.f);
    }
    if (D) {
      float* d = D + (long)z * sD;
      d[(long)(mA + i) * ldd + nA] = x00;
      d[(long)(mA + i) * ldd + nB] = x01;
      d[(long)(mB + i) * ldd + nA] = x10;
      d[(long)(mB + i) * ldd + nB] = x11;
    }
    if (Dh) {
      _Float16* d = Dh + (long)z * sDh;
      d[(long)(mA + i) * lddh + nA] = (_Float16)x00;
      d[(long)(mA + i) * lddh + nB] = (_Float16)x01;
      d[(long)(mB + i) * lddh + nA] = (_Float16)x10;
      d[(long)(mB + i) * lddh + nB] = (_Float16)x11;
    }
  }
}

// ---------------------------------------------------------------------------
// Scores GEMM: per (b,h) scores = (qn . kn^T) * exp(min(ls[h], ln100)) + b_emb
// qn: [B,H,NT,64] f16, kn: [B,H,NS,64] f16 (Bt layout), K=64 (fully unrolled).
// 32x32 tile per wave (8 WMMAs); writes f32 straight into d_out att region.
// ---------------------------------------------------------------------------
__global__ __launch_bounds__(128) void k_scores(
    const _Float16* __restrict__ qn, const _Float16* __restrict__ kn,
    const float* __restrict__ logit_scale, const float* __restrict__ bemb,
    float* __restrict__ scores) {
  const int lane = threadIdx.x & 31;
  const int wave = threadIdx.x >> 5;
  const int z = blockIdx.z;             // b*H + h
  const int b = z >> 3, h = z & 7;
  const int n0 = blockIdx.x * 32;                // key tile
  const int m0 = blockIdx.y * 128 + wave * 32;   // query tile
  const int r = lane & 15, kg = lane >> 4;

  const _Float16* ap0 = qn + (long)z * (NT * HD) + (long)(m0 + r) * HD + kg * 8;
  const _Float16* ap1 = ap0 + 16 * HD;
  const _Float16* bp0 = kn + (long)z * (NS * HD) + (long)(n0 + r) * HD + kg * 8;
  const _Float16* bp1 = bp0 + 16 * HD;

  v8f c00 = {}, c01 = {}, c10 = {}, c11 = {};
#pragma unroll
  for (int k0 = 0; k0 < HD; k0 += 32) {
    v16h a0 = load_frag(ap0 + k0);
    v16h a1 = load_frag(ap1 + k0);
    v16h b0 = load_frag(bp0 + k0);
    v16h b1 = load_frag(bp1 + k0);
    c00 = wmma(a0, b0, c00);
    c01 = wmma(a0, b1, c01);
    c10 = wmma(a1, b0, c10);
    c11 = wmma(a1, b1, c11);
  }

  const float scale = __expf(fminf(logit_scale[h], 4.6051702f)); // ln(100)
  const int mA = m0 + kg * 8, mB = mA + 16;
  const int nA = n0 + r,      nB = nA + 16;
#pragma unroll
  for (int i = 0; i < 8; ++i) {
    const long rowA = (long)b * NT + mA + i;
    const long rowB = (long)b * NT + mB + i;
    scores[((long)z * NT + mA + i) * NS + nA] =
        c00[i] * scale + bemb[(rowA * NS + nA) * HH + h];
    scores[((long)z * NT + mA + i) * NS + nB] =
        c01[i] * scale + bemb[(rowA * NS + nB) * HH + h];
    scores[((long)z * NT + mB + i) * NS + nA] =
        c10[i] * scale + bemb[(rowB * NS + nA) * HH + h];
    scores[((long)z * NT + mB + i) * NS + nB] =
        c11[i] * scale + bemb[(rowB * NS + nB) * HH + h];
  }
}

// ---------------------------------------------------------------------------
// Block-wide (256 thr = 8 wave32) reductions
// ---------------------------------------------------------------------------
__device__ inline float block_sum256(float v, float* red) {
#pragma unroll
  for (int d = 16; d; d >>= 1) v += __shfl_xor(v, d, 32);
  if ((threadIdx.x & 31) == 0) red[threadIdx.x >> 5] = v;
  __syncthreads();
  float s = 0.f;
#pragma unroll
  for (int i = 0; i < 8; ++i) s += red[i];
  __syncthreads();
  return s;
}
__device__ inline float block_max256(float v, float* red) {
#pragma unroll
  for (int d = 16; d; d >>= 1) v = fmaxf(v, __shfl_xor(v, d, 32));
  if ((threadIdx.x & 31) == 0) red[threadIdx.x >> 5] = v;
  __syncthreads();
  float s = -3.0e38f;
#pragma unroll
  for (int i = 0; i < 8; ++i) s = fmaxf(s, red[i]);
  __syncthreads();
  return s;
}

// ---------------------------------------------------------------------------
// Softmax over NS=1024, in place in d_out att region; also store f16 copy.
// ---------------------------------------------------------------------------
__global__ __launch_bounds__(256) void k_softmax(float* __restrict__ att,
                                                 _Float16* __restrict__ att_h) {
  __shared__ float red[8];
  const long row = blockIdx.x;
  float* p = att + row * (long)NS;
  const int t = threadIdx.x;
  float v0 = p[t], v1 = p[t + 256], v2 = p[t + 512], v3 = p[t + 768];
  float m = block_max256(fmaxf(fmaxf(v0, v1), fmaxf(v2, v3)), red);
  v0 = __expf(v0 - m); v1 = __expf(v1 - m);
  v2 = __expf(v2 - m); v3 = __expf(v3 - m);
  const float s = block_sum256(v0 + v1 + v2 + v3, red);
  const float inv = 1.0f / s;
  v0 *= inv; v1 *= inv; v2 *= inv; v3 *= inv;
  p[t] = v0; p[t + 256] = v1; p[t + 512] = v2; p[t + 768] = v3;
  _Float16* ph = att_h + row * (long)NS;
  ph[t] = (_Float16)v0; ph[t + 256] = (_Float16)v1;
  ph[t + 512] = (_Float16)v2; ph[t + 768] = (_Float16)v3;
}

// ---------------------------------------------------------------------------
// Per-head L2 norm of q/k: X f32 [B*Npos, 512] -> Out f16 [B,H,Npos,64].
// ---------------------------------------------------------------------------
__global__ __launch_bounds__(128) void k_headnorm(const float* __restrict__ X,
                                                  _Float16* __restrict__ Out,
                                                  int Npos) {
  const int lane = threadIdx.x & 31;
  const int wave = threadIdx.x >> 5;
  const long idx = (long)blockIdx.x * 4 + wave;   // over (B*Npos)*H
  const long row = idx >> 3;
  const int h = (int)(idx & 7);
  const float* p = X + row * 512 + h * 64;
  const float a = p[lane], b = p[lane + 32];
  float ss = a * a + b * b;
#pragma unroll
  for (int d = 16; d; d >>= 1) ss += __shfl_xor(ss, d, 32);
  const float rs = rsqrtf(ss + 1e-12f);
  const long bb = row / Npos, tpos = row % Npos;
  _Float16* o = Out + (((long)bb * HH + h) * Npos + tpos) * 64;
  o[lane] = (_Float16)(a * rs);
  o[lane + 32] = (_Float16)(b * rs);
}

// v f32 [B,Npos,512] -> vT f16 [B,H,64,Npos]  (Bt layout for ctx GEMM)
__global__ __launch_bounds__(256) void k_prep_vT(const float* __restrict__ V,
                                                 _Float16* __restrict__ vT,
                                                 int Npos) {
  long i = (long)blockIdx.x * 256 + threadIdx.x;   // over B*Npos*512
  if (i >= (long)BB * Npos * 512) return;
  const long row = i >> 9;
  const int col = (int)(i & 511);
  const int h = col >> 6, d = col & 63;
  const long b = row / Npos, s = row % Npos;
  vT[(((long)b * HH + h) * 64 + d) * Npos + s] = (_Float16)V[i];
}

// ctx_bh f32 [B,H,NT,64] -> ctx_h f16 [B,NT,512]
__global__ __launch_bounds__(256) void k_merge_ctx(const float* __restrict__ cbh,
                                                   _Float16* __restrict__ ch) {
  long i = (long)blockIdx.x * 256 + threadIdx.x;   // over B*H*NT*64
  if (i >= (long)BB * HH * NT * 64) return;
  const int d = (int)(i & 63);
  long r = i >> 6;
  const long t = r % NT; r /= NT;
  const int h = (int)(r & 7);
  const long b = r >> 3;
  ch[((b * NT + t) << 9) + h * 64 + d] = (_Float16)cbh[i];
}

// ---------------------------------------------------------------------------
// Continuous RPE MLP: rel_coords [B,NT,NS,2] -> rel_emb [.,RH], b_emb [.,H]
// ---------------------------------------------------------------------------
__global__ __launch_bounds__(256) void k_rpe(const float* __restrict__ rc,
                                             const float* __restrict__ W1,
                                             const float* __restrict__ b1,
                                             const float* __restrict__ W2,
                                             float* __restrict__ rel_emb,
                                             float* __restrict__ b_emb) {
  long i = (long)blockIdx.x * 256 + threadIdx.x;   // over B*NT*NS
  if (i >= (long)BB * NT * NS) return;
  const float c0 = rc[2 * i], c1 = rc[2 * i + 1];
  float e[RH];
#pragma unroll
  for (int j = 0; j < RH; ++j) {
    e[j] = fmaxf(fmaf(c0, W1[j], fmaf(c1, W1[RH + j], b1[j])), 0.0f);
    rel_emb[i * RH + j] = e[j];
  }
#pragma unroll
  for (int h = 0; h < HH; ++h) {
    float s = 0.0f;
#pragma unroll
    for (int j = 0; j < RH; ++j) s = fmaf(e[j], W2[j * HH + h], s);
    b_emb[i * HH + h] = s;
  }
}

// ---------------------------------------------------------------------------
// Residual + LayerNorm over 512 features; one block per row; dual stores.
// ---------------------------------------------------------------------------
__global__ __launch_bounds__(256) void k_residual_ln(
    const float* __restrict__ A, const float* __restrict__ Bv,
    const float* __restrict__ g, const float* __restrict__ be,
    float* __restrict__ Of, _Float16* __restrict__ Oh) {
  __shared__ float red[8];
  const long row = blockIdx.x;
  const int t = threadIdx.x;
  const float* pa = A + (row << 9);
  const float* pb = Bv + (row << 9);
  float x0 = pa[t] + pb[t];
  float x1 = pa[t + 256] + pb[t + 256];
  const float mu = block_sum256(x0 + x1, red) * (1.0f / 512.0f);
  const float d0 = x0 - mu, d1 = x1 - mu;
  const float var = block_sum256(d0 * d0 + d1 * d1, red) * (1.0f / 512.0f);
  const float rs = rsqrtf(var + 1e-5f);
  const float y0 = d0 * rs * g[t] + be[t];
  const float y1 = d1 * rs * g[t + 256] + be[t + 256];
  if (Of) { Of[(row << 9) + t] = y0; Of[(row << 9) + t + 256] = y1; }
  if (Oh) { Oh[(row << 9) + t] = (_Float16)y0; Oh[(row << 9) + t + 256] = (_Float16)y1; }
}

// ---------------------------------------------------------------------------
// Host orchestration
// ---------------------------------------------------------------------------
static inline char* bump(char*& p, size_t bytes) {
  char* r = p;
  p += (bytes + 255) & ~(size_t)255;
  return r;
}

extern "C" void kernel_launch(void* const* d_in, const int* in_sizes, int n_in,
                              void* d_out, int out_size, void* d_ws, size_t ws_size,
                              hipStream_t stream) {
  (void)in_sizes; (void)n_in; (void)out_size; (void)ws_size;
  const float* xs   = (const float*)d_in[0];
  const float* xt   = (const float*)d_in[1];
  const float* rc   = (const float*)d_in[2];
  const float* Wq   = (const float*)d_in[3];
  const float* bq   = (const float*)d_in[4];
  const float* Wk   = (const float*)d_in[5];
  const float* bk   = (const float*)d_in[6];
  const float* Wv   = (const float*)d_in[7];
  const float* bv   = (const float*)d_in[8];
  const float* Wo   = (const float*)d_in[9];
  const float* bo   = (const float*)d_in[10];
  const float* ls   = (const float*)d_in[11];
  const float* rW1  = (const float*)d_in[12];
  const float* rb1  = (const float*)d_in[13];
  const float* rW2  = (const float*)d_in[14];
  const float* ln1g = (const float*)d_in[15];
  const float* ln1b = (const float*)d_in[16];
  const float* mW1  = (const float*)d_in[17];
  const float* mb1  = (const float*)d_in[18];
  const float* mW2  = (const float*)d_in[19];
  const float* mb2  = (const float*)d_in[20];
  const float* ln2g = (const float*)d_in[21];
  const float* ln2b = (const float*)d_in[22];
  float* out = (float*)d_out;

  const long MQ = (long)BB * NT;        // 4096 rows of xt
  const long MS = (long)BB * NS;        // 4096 rows of xs

  char* ws = (char*)d_ws;
  _Float16* xt_h  = (_Float16*)bump(ws, MQ * EE * 2);
  _Float16* xs_h  = (_Float16*)bump(ws, MS * DD * 2);
  _Float16* WqT   = (_Float16*)bump(ws, (size_t)EE * DD * 2);
  _Float16* WkT   = (_Float16*)bump(ws, (size_t)DD * DD * 2);
  _Float16* WvT   = (_Float16*)bump(ws, (size_t)DD * DD * 2);
  _Float16* WoT   = (_Float16*)bump(ws, (size_t)DD * EE * 2);
  _Float16* W1T   = (_Float16*)bump(ws, (size_t)EE * FF * 2);
  _Float16* W2T   = (_Float16*)bump(ws, (size_t)FF * EE * 2);
  float*    qf    = (float*)bump(ws, MQ * DD * 4);
  float*    kf    = (float*)bump(ws, MS * DD * 4);
  float*    vf    = (float*)bump(ws, MS * DD * 4);
  _Float16* qn_h  = (_Float16*)bump(ws, MQ * DD * 2);
  _Float16* kn_h  = (_Float16*)bump(ws, MS * DD * 2);
  _Float16* vT_h  = (_Float16*)bump(ws, MS * DD * 2);
  _Float16* att_h = (_Float16*)bump(ws, (long)BB * HH * NT * NS * 2);
  float*    ctxbh = (float*)bump(ws, (long)BB * HH * NT * HD * 4);
  _Float16* ctx_h = (_Float16*)bump(ws, MQ * DD * 2);
  float*    aout  = (float*)bump(ws, MQ * EE * 4);
  float*    x1f   = (float*)bump(ws, MQ * EE * 4);
  _Float16* x1h   = (_Float16*)bump(ws, MQ * EE * 2);
  _Float16* hid_h = (_Float16*)bump(ws, MQ * FF * 2);
  float*    mlpf  = (float*)bump(ws, MQ * EE * 4);

  // --- stage inputs / weights to f16 ---
  k_f2h<<<(MQ * EE + 255) / 256, 256, 0, stream>>>(xt, xt_h, MQ * EE);
  k_f2h<<<(MS * DD + 255) / 256, 256, 0, stream>>>(xs, xs_h, MS * DD);
  k_transpose_f2h<<<(EE * DD + 255) / 256, 256, 0, stream>>>(Wq, WqT, EE, DD);
  k_transpose_f2h<<<(DD * DD + 255) / 256, 256, 0, stream>>>(Wk, WkT, DD, DD);
  k_transpose_f2h<<<(DD * DD + 255) / 256, 256, 0, stream>>>(Wv, WvT, DD, DD);
  k_transpose_f2h<<<(DD * EE + 255) / 256, 256, 0, stream>>>(Wo, WoT, DD, EE);
  k_transpose_f2h<<<((long)EE * FF + 255) / 256, 256, 0, stream>>>(mW1, W1T, EE, FF);
  k_transpose_f2h<<<((long)FF * EE + 255) / 256, 256, 0, stream>>>(mW2, W2T, FF, EE);

  // --- Q/K/V projections (WMMA, LDS-staged B) ---
  {
    dim3 g(DD / 32, (unsigned)(MQ / 128), 1);
    k_gemm_f16<<<g, 128, 0, stream>>>(xt_h, 0, EE, WqT, 0, EE,
                                      qf, 0, DD, nullptr, 0, 0,
                                      bq, (int)MQ, DD, EE, 0);
    k_gemm_f16<<<g, 128, 0, stream>>>(xs_h, 0, DD, WkT, 0, DD,
                                      kf, 0, DD, nullptr, 0, 0,
                                      bk, (int)MS, DD, DD, 0);
    k_gemm_f16<<<g, 128, 0, stream>>>(xs_h, 0, DD, WvT, 0, DD,
                                      vf, 0, DD, nullptr, 0, 0,
                                      bv, (int)MS, DD, DD, 0);
  }

  // --- cosine normalization, v transpose ---
  k_headnorm<<<(unsigned)(MQ * HH / 4), 128, 0, stream>>>(qf, qn_h, NT);
  k_headnorm<<<(unsigned)(MS * HH / 4), 128, 0, stream>>>(kf, kn_h, NS);
  k_prep_vT<<<(unsigned)((MS * DD + 255) / 256), 256, 0, stream>>>(vf, vT_h, NS);

  // --- RPE MLP -> rel_emb, b_emb (direct to d_out) ---
  k_rpe<<<(unsigned)(((long)BB * NT * NS + 255) / 256), 256, 0, stream>>>(
      rc, rW1, rb1, rW2, out + OFF_RELEMB, out + OFF_BEMB);

  // --- scores (WMMA) straight into d_out att region ---
  {
    dim3 g(NS / 32, NT / 128, BB * HH);
    k_scores<<<g, 128, 0, stream>>>(qn_h, kn_h, ls, out + OFF_BEMB, out + OFF_ATT);
  }

  // --- softmax in place + f16 copy ---
  k_softmax<<<(unsigned)((long)BB * HH * NT), 256, 0, stream>>>(out + OFF_ATT, att_h);

  // --- ctx = att @ v (WMMA, batched over B*H) ---
  {
    dim3 g(HD / 32, NT / 128, BB * HH);
    k_gemm_f16<<<g, 128, 0, stream>>>(att_h, (long)NT * NS, NS,
                                      vT_h, (long)HD * NS, NS,
                                      ctxbh, (long)NT * HD, HD, nullptr, 0, 0,
                                      nullptr, NT, HD, NS, 0);
  }
  k_merge_ctx<<<(unsigned)(((long)BB * HH * NT * HD + 255) / 256), 256, 0, stream>>>(
      ctxbh, ctx_h);

  // --- output projection (WMMA) ---
  {
    dim3 g(EE / 32, (unsigned)(MQ / 128), 1);
    k_gemm_f16<<<g, 128, 0, stream>>>(ctx_h, 0, DD, WoT, 0, DD,
                                      aout, 0, EE, nullptr, 0, 0,
                                      bo, (int)MQ, EE, DD, 0);
  }

  // --- residual + LN1 ---
  k_residual_ln<<<(unsigned)MQ, 256, 0, stream>>>(xt, aout, ln1g, ln1b, x1f, x1h);

  // --- MLP (WMMA): relu(x@W1+b1) -> relu(.@W2+b2) ---
  {
    dim3 g1(FF / 32, (unsigned)(MQ / 128), 1);
    k_gemm_f16<<<g1, 128, 0, stream>>>(x1h, 0, EE, W1T, 0, EE,
                                       nullptr, 0, 0, hid_h, 0, FF,
                                       mb1, (int)MQ, FF, EE, 1);
    dim3 g2(EE / 32, (unsigned)(MQ / 128), 1);
    k_gemm_f16<<<g2, 128, 0, stream>>>(hid_h, 0, FF, W2T, 0, FF,
                                       mlpf, 0, EE, nullptr, 0, 0,
                                       mb2, (int)MQ, EE, FF, 1);
  }

  // --- residual + LN2 -> x output region ---
  k_residual_ln<<<(unsigned)MQ, 256, 0, stream>>>(x1f, mlpf, ln2g, ln2b,
                                                  out + OFF_X, nullptr);
}
